// GlobalConcepts_56470230008398
// MI455X (gfx1250) — matrix operand order repaired and checked by reading, not compiled, not run on hardware
//
#include <hip/hip_runtime.h>
#include <math.h>

typedef __attribute__((ext_vector_type(16))) _Float16 v16h;
typedef __attribute__((ext_vector_type(8)))  float    v8f;

namespace {
constexpr int   kK       = 1024;   // mixture components
constexpr int   kD       = 64;     // feature dim
constexpr int   kBS      = 65536;  // B*S rows
constexpr int   kThreads = 256;    // 8 waves (wave32)
constexpr int   kNPart   = 64;     // partial buffers for responsibility sums
constexpr float kLog2Pi  = 1.8378770664093454f;
constexpr float kDecay   = 0.9f;

// workspace byte offsets
constexpr size_t kWsW    = 0;        // swizzled W: 128*1024 f16 = 262144 B
constexpr size_t kWsBias = 262144;   // 1024 f32
constexpr size_t kWsSig  = 266240;   // 65536 f32 (sigma = exp(log_sigma))
constexpr size_t kWsRsum = 528384;   // 64*1024 f32 partial responsibility sums
}

// Map W element (row j in 0..127 of the augmented K-dim, column k in 0..1023)
// to the flat index giving the 32x16 f16 B-operand layout in the k order this code uses (same as A):
// chunk c = j/32 covers 32 K-rows; within a chunk, lanes 0-15 hold rows 0-15
// (element e of v16h -> row e), lanes 16-31 hold rows 16-31; lane%16 = column.
__device__ __forceinline__ int w_swizzle(int j, int k) {
  int t = k >> 4, n = k & 15;          // column tile / column-in-tile
  int c = j >> 5, r = j & 31;          // K-chunk / row-in-chunk
  int g = r >> 4, q = r & 15;          // lane half / v16h element
  int lane = g * 16 + n;
  return (((t * 4 + c) * 32 + lane) << 4) + q;
}

// ---------------- prep 1: per (k,d) weights + sigma; zero partials ----------
__global__ __launch_bounds__(256) void gmm_prep1(
    const float* __restrict__ mu, const float* __restrict__ log_sigma,
    _Float16* __restrict__ W, float* __restrict__ sigma_ws,
    float* __restrict__ rsum_part) {
  int tid = blockIdx.x * 256 + threadIdx.x;   // 0 .. 65535
  int k = tid >> 6, d = tid & 63;
  float ls = log_sigma[tid];
  float m  = mu[tid];
  float iv = __expf(-2.0f * ls);
  sigma_ws[tid] = __expf(ls);
  rsum_part[tid] = 0.f;                       // 64*1024 == 65536 entries
  // linear coefficient row j=d : mu*inv_var ; quadratic row j=64+d : -0.5*inv_var
  W[w_swizzle(d, k)]      = (_Float16)(m * iv);
  W[w_swizzle(64 + d, k)] = (_Float16)(-0.5f * iv);
}

// ---------------- prep 2: per-k bias ----------------
__global__ __launch_bounds__(256) void gmm_prep2(
    const float* __restrict__ mu, const float* __restrict__ log_sigma,
    float* __restrict__ bias) {
  int k = blockIdx.x * 256 + threadIdx.x;     // 0 .. 1023
  float s_m2 = 0.f, s_ls = 0.f;
  for (int d = 0; d < kD; ++d) {
    float ls = log_sigma[k * kD + d];
    float m  = mu[k * kD + d];
    s_m2 += m * m * __expf(-2.0f * ls);
    s_ls += ls;
  }
  bias[k] = -0.5f * s_m2 - s_ls - 0.5f * (float)kD * kLog2Pi;
}

// ---------------- main fused kernel: 16 rows per workgroup ----------------
// 8 waves; wave w owns column tiles t = w*8 .. w*8+7. The whole 16x1024
// log-prob tile stays in accumulator registers (acc[8] per lane); LDS is used
// only for slot staging and small padded two-stage reductions.
__global__ __launch_bounds__(kThreads) void gmm_main(
    const float* __restrict__ slots, const float* __restrict__ mu,
    const float* __restrict__ z, const _Float16* __restrict__ W,
    const float* __restrict__ bias, const float* __restrict__ sigma,
    float* __restrict__ rsum_part, float* __restrict__ out_slots,
    float* __restrict__ out_ll) {
  __shared__ float s_slots[16 * kD];      // staged rows
  __shared__ float s_candf[128 * 17];     // [slot][row], pad 17 (conflict-free)
  __shared__ int   s_candi[128 * 17];
  __shared__ float s_p2f[16 * 17];        // stage-2 partials [sgroup][row]
  __shared__ int   s_p2i[16 * 17];
  __shared__ float s_rowmax[16];
  __shared__ int   s_rowarg[16];
  __shared__ float s_rowsum[16];

  const int tid  = threadIdx.x;
  const int lane = tid & 31;
  const int w    = tid >> 5;              // wave 0..7
  const int n    = lane & 15;             // column-in-tile
  const int g    = lane >> 4;             // row-half selector
  const int row0 = blockIdx.x * 16;
  const int slot = w * 16 + n;            // 0..127 candidate slot

  // ---- stage 16 rows of slots (coalesced 4 KB) ----
  for (int i = tid; i < 16 * kD; i += kThreads) s_slots[i] = slots[row0 * kD + i];
  __syncthreads();

  // ---- build A operand: 16x128 f16, 4 chunks of 16x32 ----
  // lane holds row m = lane&15; element i of chunk c -> K index
  // 32*c + i + 16*g. j<64 -> slots ; j>=64 -> slots^2.
  const int m = lane & 15;
  v16h a[4];
#pragma unroll
  for (int c = 0; c < 4; ++c) {
#pragma unroll
    for (int i = 0; i < 16; ++i) {
      int j = 32 * c + i + 16 * g;
      float v;
      if (j < 64) {
        v = s_slots[m * kD + j];
      } else {
        float x = s_slots[m * kD + (j - 64)];
        v = x * x;
      }
      a[c][i] = (_Float16)v;
    }
  }

  // ---- GEMM: 8 tiles per wave, log-probs kept in registers ----
  v8f acc[8];
#pragma unroll
  for (int s = 0; s < 8; ++s) {
    const int t = w * 8 + s;
    v8f c = {};
#pragma unroll
    for (int ch = 0; ch < 4; ++ch) {
      v16h b = *(const v16h*)(W + (((t * 4 + ch) * 32 + lane) << 4));
      c = __builtin_amdgcn_wmma_f32_16x16x32_f16(
          /*neg_a=*/false, a[ch], /*neg_b=*/false, b,
          /*c_mod=*/(short)0, c, /*reuse_a=*/false, /*reuse_b=*/false);
    }
    float bn = bias[t * 16 + n];            // bias depends only on column
#pragma unroll
    for (int r = 0; r < 8; ++r) c[r] += bn;
    acc[s] = c;
  }

  // ---- per-lane row max / argmax over the wave's 8 tiles ----
  {
    float rmax[8];
    int   rarg[8];
#pragma unroll
    for (int r = 0; r < 8; ++r) { rmax[r] = -INFINITY; rarg[r] = 0x7fffffff; }
#pragma unroll
    for (int s = 0; s < 8; ++s) {
      const int col = (w * 8 + s) * 16 + n;
#pragma unroll
      for (int r = 0; r < 8; ++r) {
        float v = acc[s][r];
        if (v > rmax[r]) { rmax[r] = v; rarg[r] = col; }
      }
    }
#pragma unroll
    for (int r = 0; r < 8; ++r) {
      s_candf[slot * 17 + (r + 8 * g)] = rmax[r];
      s_candi[slot * 17 + (r + 8 * g)] = rarg[r];
    }
  }
  __syncthreads();
  // stage 1: 256 threads reduce 8 slots each (conflict-free, parallel)
  {
    const int row = tid & 15, sg = tid >> 4;
    float mx = -INFINITY; int am = 0x7fffffff;
    for (int q = 0; q < 8; ++q) {
      int sl = sg * 8 + q;
      float v = s_candf[sl * 17 + row];
      int  ai = s_candi[sl * 17 + row];
      if (v > mx || (v == mx && ai < am)) { mx = v; am = ai; }
    }
    s_p2f[sg * 17 + row] = mx;
    s_p2i[sg * 17 + row] = am;
  }
  __syncthreads();
  if (tid < 16) {                           // stage 2: 16 partials per row
    float mx = -INFINITY; int am = 0x7fffffff;
    for (int sg = 0; sg < 16; ++sg) {
      float v = s_p2f[sg * 17 + tid];
      int  ai = s_p2i[sg * 17 + tid];
      if (v > mx || (v == mx && ai < am)) { mx = v; am = ai; }
    }
    s_rowmax[tid] = mx;
    s_rowarg[tid] = am;
    out_ll[row0 + tid] = mx;                // log-likelihood = lp[argmax] = rowmax
  }
  __syncthreads();

  // ---- exp in registers + row sums ----
  {
    float rm[8], rsum[8];
#pragma unroll
    for (int r = 0; r < 8; ++r) { rm[r] = s_rowmax[r + 8 * g]; rsum[r] = 0.f; }
#pragma unroll
    for (int s = 0; s < 8; ++s) {
#pragma unroll
      for (int r = 0; r < 8; ++r) {
        float e = __expf(acc[s][r] - rm[r]);
        acc[s][r] = e;
        rsum[r] += e;
      }
    }
#pragma unroll
    for (int r = 0; r < 8; ++r)
      s_candf[slot * 17 + (r + 8 * g)] = rsum[r];
  }
  __syncthreads();
  {
    const int row = tid & 15, sg = tid >> 4;
    float s = 0.f;
    for (int q = 0; q < 8; ++q) s += s_candf[(sg * 8 + q) * 17 + row];
    s_p2f[sg * 17 + row] = s;
  }
  __syncthreads();
  if (tid < 16) {
    float s = 0.f;
    for (int sg = 0; sg < 16; ++sg) s += s_p2f[sg * 17 + tid];
    s_rowsum[tid] = s;
  }
  __syncthreads();

  // ---- per-column responsibility sums -> spread atomic partials ----
  {
    float* part = rsum_part + (size_t)(blockIdx.x & (kNPart - 1)) * kK;
    float rinv[8];
#pragma unroll
    for (int r = 0; r < 8; ++r) rinv[r] = 1.f / s_rowsum[r + 8 * g];
#pragma unroll
    for (int s = 0; s < 8; ++s) {
      float cs = 0.f;
#pragma unroll
      for (int r = 0; r < 8; ++r) cs += acc[s][r] * rinv[r];
      cs += __shfl_xor(cs, 16, 32);         // combine row halves (lane ^ 16)
      if (g == 0) atomicAdd(&part[(w * 8 + s) * 16 + n], cs);
    }
  }

  // ---- new_slots = mu[arg] + sigma[arg] * z ----
  for (int i = tid; i < 16 * kD; i += kThreads) {
    int r = i >> 6, d = i & 63;
    int aidx = s_rowarg[r];
    out_slots[row0 * kD + i] = mu[aidx * kD + d] + sigma[aidx * kD + d] * z[row0 * kD + i];
  }
}

// ---------------- finalize: EMA of the two softmaxes over K ----------------
__global__ __launch_bounds__(256) void gmm_finalize(
    const float* __restrict__ log_prior, const float* __restrict__ rsum_part,
    float* __restrict__ out_prior) {
  __shared__ float red[256];
  __shared__ float s_rs[kK];
  const int tid = threadIdx.x;

  // total responsibility sums = sum of the 64 partial buffers
  for (int k = tid; k < kK; k += 256) {
    float t = 0.f;
    for (int p = 0; p < kNPart; ++p) t += rsum_part[p * kK + k];
    s_rs[k] = t;
  }
  __syncthreads();

  float m1 = -INFINITY, m2 = -INFINITY;
  for (int k = tid; k < kK; k += 256) {
    m1 = fmaxf(m1, log_prior[k]);
    m2 = fmaxf(m2, s_rs[k]);
  }
  red[tid] = m1; __syncthreads();
  for (int s = 128; s > 0; s >>= 1) {
    if (tid < s) red[tid] = fmaxf(red[tid], red[tid + s]);
    __syncthreads();
  }
  m1 = red[0]; __syncthreads();
  red[tid] = m2; __syncthreads();
  for (int s = 128; s > 0; s >>= 1) {
    if (tid < s) red[tid] = fmaxf(red[tid], red[tid + s]);
    __syncthreads();
  }
  m2 = red[0]; __syncthreads();

  float s1 = 0.f, s2 = 0.f;
  for (int k = tid; k < kK; k += 256) {
    s1 += __expf(log_prior[k] - m1);
    s2 += __expf(s_rs[k] - m2);
  }
  red[tid] = s1; __syncthreads();
  for (int s = 128; s > 0; s >>= 1) {
    if (tid < s) red[tid] += red[tid + s];
    __syncthreads();
  }
  s1 = red[0]; __syncthreads();
  red[tid] = s2; __syncthreads();
  for (int s = 128; s > 0; s >>= 1) {
    if (tid < s) red[tid] += red[tid + s];
    __syncthreads();
  }
  s2 = red[0]; __syncthreads();

  float i1 = 1.f / s1, i2 = 1.f / s2;
  for (int k = tid; k < kK; k += 256)
    out_prior[k] = kDecay * __expf(log_prior[k] - m1) * i1 +
                   (1.f - kDecay) * __expf(s_rs[k] - m2) * i2;
}

extern "C" void kernel_launch(void* const* d_in, const int* in_sizes, int n_in,
                              void* d_out, int out_size, void* d_ws, size_t ws_size,
                              hipStream_t stream) {
  (void)in_sizes; (void)n_in; (void)out_size; (void)ws_size;
  const float* slots     = (const float*)d_in[0];
  const float* mu        = (const float*)d_in[1];
  const float* log_sigma = (const float*)d_in[2];
  const float* log_prior = (const float*)d_in[3];
  const float* z         = (const float*)d_in[4];

  float* out       = (float*)d_out;
  float* out_slots = out;                      // [B,S,D] = 4194304
  float* out_ll    = out + (size_t)kBS * kD;   // [B,S]   = 65536
  float* out_prior = out_ll + kBS;             // [K]     = 1024

  char* ws = (char*)d_ws;
  _Float16* W  = (_Float16*)(ws + kWsW);
  float* bias  = (float*)(ws + kWsBias);
  float* sigw  = (float*)(ws + kWsSig);
  float* rpart = (float*)(ws + kWsRsum);

  gmm_prep1<<<kK * kD / 256, 256, 0, stream>>>(mu, log_sigma, W, sigw, rpart);
  gmm_prep2<<<kK / 256, 256, 0, stream>>>(mu, log_sigma, bias);
  gmm_main<<<kBS / 16, kThreads, 0, stream>>>(
      slots, mu, z, W, bias, sigw, rpart, out_slots, out_ll);
  gmm_finalize<<<1, 256, 0, stream>>>(log_prior, rpart, out_prior);
}
